// GeometricTreeEncoder_33509334843753
// MI455X (gfx1250) — matrix operand order repaired and checked by reading, not compiled, not run on hardware
//
#include <hip/hip_runtime.h>
#include <hip/hip_bf16.h>

// ---------------------------------------------------------------------------
// GeometricTreeEncoder forward for MI455X (gfx1250), wave32 + WMMA f16->f32.
//
// Input order (jax pytree, dict keys sorted):
//  0 x[N*3] 1 pos[N*3] 2 edge_index[2*E] 3 batch[N]
//  4 input_proj.b[64] 5 input_proj.w[3*64]
//  per layer l (base = 6 + 10*l):
//    +0 ln_b[64] +1 ln_g[64] +2 msg1.b[64] +3 msg1.w[134*64]
//    +4 msg2.b[64] +5 msg2.w[64*64] +6 upd1.b[64] +7 upd1.w[128*64]
//    +8 upd2.b[64] +9 upd2.w[64*64]
//  36 output_proj.b[128] 37 output_proj.w[64*128]
//  38 pool1.b[64] 39 pool1.w[128*64] 40 pool2.b[128] 41 pool2.w[64*128]
// ---------------------------------------------------------------------------

#define NN   50000
#define EE   1000000
#define HH   64
#define GG   64
#define OUTD 128

typedef _Float16 v16h __attribute__((ext_vector_type(16)));
typedef _Float16 v8h  __attribute__((ext_vector_type(8)));
typedef _Float16 v4h  __attribute__((ext_vector_type(4)));
typedef float    v8f  __attribute__((ext_vector_type(8)));

__device__ __forceinline__ v8f zero8() {
  v8f z = {0.f,0.f,0.f,0.f,0.f,0.f,0.f,0.f};
  return z;
}

// A-fragment (16x32 f16, M=row within tile). ISA 7.12.2:
// lane l: m = l&15, hi = l>>4; halves 0..7 -> k = k0+hi*8+t,
// halves 8..15 -> k = k0+16+hi*8+(t-8). Two contiguous 16B runs.
__device__ __forceinline__ v16h load_a_frag(const _Float16* tile, int rstride,
                                            int m, int hi, int k0) {
  const _Float16* p = tile + m * rstride + k0 + hi * 8;
  v8h lo = *(const v8h*)(p);
  v8h hh = *(const v8h*)(p + 16);
  v16h a;
#pragma unroll
  for (int i = 0; i < 8; ++i) { a[i] = lo[i]; a[8 + i] = hh[i]; }
  return a;
}

// B-fragment (32x16 f16, K x N), weights stored col-major [n][Kpad]:
// lane l: n = ntile*16 + (l&15), hi = l>>4; halves 0..7 -> k = k0+hi*8+t,
// halves 8..15 -> k = k0+16+hi*8+(t-8).
__device__ __forceinline__ v16h load_b_frag(const _Float16* wt, int Kpad,
                                            int nCol, int hi, int k0) {
  const _Float16* p = wt + nCol * Kpad + k0 + hi * 8;
  v8h lo = *(const v8h*)(p);
  v8h hh = *(const v8h*)(p + 16);
  v16h b;
#pragma unroll
  for (int i = 0; i < 8; ++i) { b[i] = lo[i]; b[8 + i] = hh[i]; }
  return b;
}

__device__ __forceinline__ v8f wmma_f16(v16h a, v16h b, v8f c) {
  return __builtin_amdgcn_wmma_f32_16x16x32_f16(false, a, false, b,
                                                (short)0, c, false, false);
}

__device__ __forceinline__ void atomic_add_f32(float* p, float v) {
  __hip_atomic_fetch_add(p, v, __ATOMIC_RELAXED, __HIP_MEMORY_SCOPE_AGENT);
}

// ---------------------------------------------------------------------------
// Pack fp32 row-major [K x Nout] weight into f16 col-major [Nout x Kpad].
// ---------------------------------------------------------------------------
__global__ void k_pack(const float* __restrict__ w, _Float16* __restrict__ wt,
                       int K, int Kpad, int Nout) {
  int i = blockIdx.x * blockDim.x + threadIdx.x;
  if (i >= Nout * Kpad) return;
  int o = i / Kpad, k = i - o * Kpad;
  wt[i] = (k < K) ? (_Float16)w[k * Nout + o] : (_Float16)0.0f;
}

// ---------------------------------------------------------------------------
// h = relu(x @ Win + b),  x:[N,3], Win:[3,64]
// ---------------------------------------------------------------------------
__global__ void k_input(const float* __restrict__ x, const float* __restrict__ w,
                        const float* __restrict__ b, float* __restrict__ h) {
  int i = blockIdx.x * blockDim.x + threadIdx.x;
  if (i >= NN * HH) return;
  int n = i >> 6, o = i & 63;
  float a = b[o] + x[n * 3 + 0] * w[o] + x[n * 3 + 1] * w[64 + o] +
            x[n * 3 + 2] * w[128 + o];
  h[i] = a > 0.f ? a : 0.f;
}

// ---------------------------------------------------------------------------
// Fused edge kernel: gather -> msg1 (WMMA) -> relu -> msg2 (WMMA) -> atomic
// scatter into aggr[dst]. 8 waves/block, 16 edges per wave.
// ---------------------------------------------------------------------------
__global__ __launch_bounds__(256) void k_edge(
    const float* __restrict__ h, const float* __restrict__ pos,
    const int* __restrict__ srcIdx, const int* __restrict__ dstIdx,
    const _Float16* __restrict__ gW1, const float* __restrict__ gB1,
    const _Float16* __restrict__ gW2, const float* __restrict__ gB2,
    float* __restrict__ aggr) {
  const int K1P = 160;   // 134 padded to 5*32
  const int AS  = 168;   // A row stride (halves), mult of 8
  const int TS  = 72;    // t row stride

  __shared__ __align__(16) _Float16 sW1[64 * 160];
  __shared__ __align__(16) _Float16 sW2[64 * 64];
  __shared__ float sB1[64], sB2[64];
  __shared__ __align__(16) _Float16 sA[8][16 * 168];
  __shared__ __align__(16) _Float16 sT[8][16 * 72];
  __shared__ int sDst[8][16];

  const int tid = threadIdx.x, lane = tid & 31, wave = tid >> 5;
  for (int i = tid; i < 64 * 160; i += 256) sW1[i] = gW1[i];
  for (int i = tid; i < 64 * 64; i += 256) sW2[i] = gW2[i];
  if (tid < 64) { sB1[tid] = gB1[tid]; sB2[tid] = gB2[tid]; }
  __syncthreads();

  const int tile = blockIdx.x * 8 + wave;
  const int e0 = tile * 16;
  const bool tvalid = e0 < EE;
  const int m = lane & 15, hi = lane >> 4, nLane = lane & 15;

  int e = e0 + m; if (e >= EE) e = EE - 1;
  const int s = srcIdx[e], d = dstIdx[e];
  _Float16* A = &sA[wave][0];

  if (hi == 0) {
    const float4* hp = (const float4*)(h + (size_t)d * HH);
#pragma unroll
    for (int c = 0; c < 16; ++c) {
      float4 v = hp[c];
      v4h t4 = {(_Float16)v.x, (_Float16)v.y, (_Float16)v.z, (_Float16)v.w};
      *(v4h*)(A + m * AS + c * 4) = t4;
    }
    sDst[wave][m] = d;
  } else {
    const float4* hp = (const float4*)(h + (size_t)s * HH);
#pragma unroll
    for (int c = 0; c < 16; ++c) {
      float4 v = hp[c];
      v4h t4 = {(_Float16)v.x, (_Float16)v.y, (_Float16)v.z, (_Float16)v.w};
      *(v4h*)(A + m * AS + 64 + c * 4) = t4;
    }
    float dx = pos[d * 3 + 0] - pos[s * 3 + 0];
    float dy = pos[d * 3 + 1] - pos[s * 3 + 1];
    float dz = pos[d * 3 + 2] - pos[s * 3 + 2];
    float sq = dx * dx + dy * dy + dz * dz;
    float dist = sqrtf(sq);
    float inv = 1.0f / (dist + 1e-8f);
    A[m * AS + 128] = (_Float16)dist;
    A[m * AS + 129] = (_Float16)(dx * inv);
    A[m * AS + 130] = (_Float16)(dy * inv);
    A[m * AS + 131] = (_Float16)(dz * inv);
    A[m * AS + 132] = (_Float16)(dist * dist);
    A[m * AS + 133] = (_Float16)sq;
#pragma unroll
    for (int k = 134; k < K1P; ++k) A[m * AS + k] = (_Float16)0.0f;
  }
  __syncthreads();

  // GEMM1: [16 x 160] x [160 x 64] -> relu -> sT
  _Float16* T = &sT[wave][0];
#pragma unroll
  for (int nt = 0; nt < 4; ++nt) {
    v8f acc = zero8();
#pragma unroll
    for (int kt = 0; kt < 5; ++kt) {
      v16h a = load_a_frag(A, AS, m, hi, kt * 32);
      v16h b = load_b_frag(sW1, K1P, nt * 16 + nLane, hi, kt * 32);
      acc = wmma_f16(a, b, acc);
    }
    const int n = nt * 16 + nLane;
    const float bia = sB1[n];
#pragma unroll
    for (int r = 0; r < 8; ++r) {
      int mr = r + hi * 8;
      float v = acc[r] + bia;
      T[mr * TS + n] = (_Float16)(v > 0.f ? v : 0.f);
    }
  }
  __syncthreads();

  // GEMM2: [16 x 64] x [64 x 64] -> atomic scatter
#pragma unroll
  for (int nt = 0; nt < 4; ++nt) {
    v8f acc = zero8();
#pragma unroll
    for (int kt = 0; kt < 2; ++kt) {
      v16h a = load_a_frag(T, TS, m, hi, kt * 32);
      v16h b = load_b_frag(sW2, 64, nt * 16 + nLane, hi, kt * 32);
      acc = wmma_f16(a, b, acc);
    }
    const int n = nt * 16 + nLane;
    const float bia = sB2[n];
    if (tvalid) {
#pragma unroll
      for (int r = 0; r < 8; ++r) {
        int mr = r + hi * 8;
        atomic_add_f32(&aggr[(size_t)sDst[wave][mr] * HH + n], acc[r] + bia);
      }
    }
  }
}

// ---------------------------------------------------------------------------
// Fused update kernel: [h|aggr] -> upd1 (WMMA) -> relu -> upd2 (WMMA)
//   -> LayerNorm -> relu -> +h residual. 16 nodes per wave, 8 waves/block.
// ---------------------------------------------------------------------------
__global__ __launch_bounds__(256) void k_update(
    const float* __restrict__ h, const float* __restrict__ aggr,
    const _Float16* __restrict__ gU1, const float* __restrict__ gB1,
    const _Float16* __restrict__ gU2, const float* __restrict__ gB2,
    const float* __restrict__ lnG, const float* __restrict__ lnB,
    float* __restrict__ hOut) {
  const int KP = 128;
  const int AS = 136;
  const int TS = 72;
  const int HS = 65;  // f32 stride -> conflict-free row walks

  __shared__ __align__(16) _Float16 sU1[64 * 128];
  __shared__ __align__(16) _Float16 sU2[64 * 64];
  __shared__ float sB1[64], sB2[64], sG[64], sBt[64];
  __shared__ __align__(16) _Float16 sA[8][16 * 136];
  __shared__ __align__(16) _Float16 sT[8][16 * 72];
  __shared__ float sH2[8][16 * 65];

  const int tid = threadIdx.x, lane = tid & 31, wave = tid >> 5;
  for (int i = tid; i < 64 * 128; i += 256) sU1[i] = gU1[i];
  for (int i = tid; i < 64 * 64; i += 256) sU2[i] = gU2[i];
  if (tid < 64) { sB1[tid] = gB1[tid]; sB2[tid] = gB2[tid];
                  sG[tid] = lnG[tid];  sBt[tid] = lnB[tid]; }
  __syncthreads();

  const int tile = blockIdx.x * 8 + wave;
  const int n0 = tile * 16;
  const int m = lane & 15, hi = lane >> 4, nLane = lane & 15;
  int node = n0 + m; if (node >= NN) node = NN - 1;
  _Float16* A = &sA[wave][0];

  {
    const float4* src = (const float4*)((hi == 0 ? h : aggr) + (size_t)node * HH);
    const int koff = hi == 0 ? 0 : 64;
#pragma unroll
    for (int c = 0; c < 16; ++c) {
      float4 v = src[c];
      v4h t4 = {(_Float16)v.x, (_Float16)v.y, (_Float16)v.z, (_Float16)v.w};
      *(v4h*)(A + m * AS + koff + c * 4) = t4;
    }
  }
  __syncthreads();

  _Float16* T = &sT[wave][0];
#pragma unroll
  for (int nt = 0; nt < 4; ++nt) {
    v8f acc = zero8();
#pragma unroll
    for (int kt = 0; kt < 4; ++kt) {
      v16h a = load_a_frag(A, AS, m, hi, kt * 32);
      v16h b = load_b_frag(sU1, KP, nt * 16 + nLane, hi, kt * 32);
      acc = wmma_f16(a, b, acc);
    }
    const int n = nt * 16 + nLane;
    const float bia = sB1[n];
#pragma unroll
    for (int r = 0; r < 8; ++r) {
      int mr = r + hi * 8;
      float v = acc[r] + bia;
      T[mr * TS + n] = (_Float16)(v > 0.f ? v : 0.f);
    }
  }
  __syncthreads();

#pragma unroll
  for (int nt = 0; nt < 4; ++nt) {
    v8f acc = zero8();
#pragma unroll
    for (int kt = 0; kt < 2; ++kt) {
      v16h a = load_a_frag(T, TS, m, hi, kt * 32);
      v16h b = load_b_frag(sU2, 64, nt * 16 + nLane, hi, kt * 32);
      acc = wmma_f16(a, b, acc);
    }
    const int n = nt * 16 + nLane;
    const float bia = sB2[n];
#pragma unroll
    for (int r = 0; r < 8; ++r) {
      int mr = r + hi * 8;
      sH2[wave][mr * HS + n] = acc[r] + bia;
    }
  }
  __syncthreads();

  // LayerNorm + relu + residual (lanes 0..15, one row each)
  if (hi == 0) {
    const int node2 = n0 + m;
    if (node2 < NN) {
      const float* row = &sH2[wave][m * HS];
      float mu = 0.f;
#pragma unroll
      for (int k = 0; k < HH; ++k) mu += row[k];
      mu *= (1.0f / HH);
      float var = 0.f;
#pragma unroll
      for (int k = 0; k < HH; ++k) { float dv = row[k] - mu; var += dv * dv; }
      var *= (1.0f / HH);
      const float rs = rsqrtf(var + 1e-5f);
#pragma unroll
      for (int k = 0; k < HH; ++k) {
        float v = (row[k] - mu) * rs * sG[k] + sBt[k];
        v = v > 0.f ? v : 0.f;
        hOut[(size_t)node2 * HH + k] = v + h[(size_t)node2 * HH + k];
      }
    }
  }
}

// ---------------------------------------------------------------------------
// out_nodes = h @ Wout + b; segment-sum into gsum[batch], counts into gcnt.
// One node per block (128 threads).
// ---------------------------------------------------------------------------
__global__ __launch_bounds__(128) void k_outproj(
    const float* __restrict__ h, const int* __restrict__ batch,
    const float* __restrict__ Wout, const float* __restrict__ Bout,
    float* __restrict__ gsum, float* __restrict__ gcnt) {
  __shared__ float sh[64];
  const int n = blockIdx.x, t = threadIdx.x;
  if (t < 64) sh[t] = h[(size_t)n * HH + t];
  __syncthreads();
  const int b = batch[n];
  float acc = Bout[t];
#pragma unroll
  for (int k = 0; k < 64; ++k) acc += sh[k] * Wout[k * OUTD + t];
  atomic_add_f32(&gsum[(size_t)b * OUTD + t], acc);
  if (t == 0) atomic_add_f32(&gcnt[b], 1.0f);
}

// ---------------------------------------------------------------------------
// means -> pool1 -> relu -> pool2. Single block.
// ---------------------------------------------------------------------------
__global__ __launch_bounds__(256) void k_pool(
    const float* __restrict__ gsum, const float* __restrict__ gcnt,
    const float* __restrict__ P1w, const float* __restrict__ P1b,
    const float* __restrict__ P2w, const float* __restrict__ P2b,
    float* __restrict__ out) {
  __shared__ float M[GG * OUTD];
  __shared__ float P[GG * HH];
  const int t = threadIdx.x;
  for (int i = t; i < GG * OUTD; i += 256) {
    int g = i >> 7;
    float c = gcnt[g];
    M[i] = (c > 0.f) ? gsum[i] / fmaxf(c, 1.0f) : 0.f;
  }
  __syncthreads();
  for (int i = t; i < GG * HH; i += 256) {
    int g = i >> 6, o = i & 63;
    float a = P1b[o];
    for (int k = 0; k < OUTD; ++k) a += M[g * OUTD + k] * P1w[k * HH + o];
    P[i] = a > 0.f ? a : 0.f;
  }
  __syncthreads();
  for (int i = t; i < GG * OUTD; i += 256) {
    int g = i >> 7, o = i & 127;
    float a = P2b[o];
    for (int k = 0; k < HH; ++k) a += P[g * HH + k] * P2w[k * OUTD + o];
    out[i] = a;
  }
}

// ---------------------------------------------------------------------------
extern "C" void kernel_launch(void* const* d_in, const int* in_sizes, int n_in,
                              void* d_out, int out_size, void* d_ws, size_t ws_size,
                              hipStream_t stream) {
  (void)in_sizes; (void)n_in; (void)out_size; (void)ws_size;

  const float* x    = (const float*)d_in[0];
  const float* pos  = (const float*)d_in[1];
  const int*   eidx = (const int*)d_in[2];
  const int*   batch = (const int*)d_in[3];
  const float* in_b = (const float*)d_in[4];
  const float* in_w = (const float*)d_in[5];
  const float* out_b = (const float*)d_in[36];
  const float* out_w = (const float*)d_in[37];
  const float* p1b = (const float*)d_in[38];
  const float* p1w = (const float*)d_in[39];
  const float* p2b = (const float*)d_in[40];
  const float* p2w = (const float*)d_in[41];

  const int* srcIdx = eidx;        // edge_index[0]
  const int* dstIdx = eidx + EE;   // edge_index[1]

  // ---- workspace carve-up (256B aligned) ----
  size_t off = 0;
  auto carve = [&](size_t bytes) -> void* {
    void* p = (char*)d_ws + off;
    off += (bytes + 255) & ~(size_t)255;
    return p;
  };
  float* hA   = (float*)carve((size_t)NN * HH * 4);
  float* hB   = (float*)carve((size_t)NN * HH * 4);
  float* aggr = (float*)carve((size_t)NN * HH * 4);
  float* gsum = (float*)carve((size_t)GG * OUTD * 4);
  float* gcnt = (float*)carve((size_t)GG * 4);
  _Float16* w1t[3]; _Float16* w2t[3]; _Float16* u1t[3]; _Float16* u2t[3];
  for (int l = 0; l < 3; ++l) {
    w1t[l] = (_Float16*)carve((size_t)64 * 160 * 2);
    w2t[l] = (_Float16*)carve((size_t)64 * 64 * 2);
    u1t[l] = (_Float16*)carve((size_t)64 * 128 * 2);
    u2t[l] = (_Float16*)carve((size_t)64 * 64 * 2);
  }

  // ---- pack weights to f16 col-major ----
  for (int l = 0; l < 3; ++l) {
    const int base = 6 + 10 * l;
    const float* m1w = (const float*)d_in[base + 3];
    const float* m2w = (const float*)d_in[base + 5];
    const float* uw1 = (const float*)d_in[base + 7];
    const float* uw2 = (const float*)d_in[base + 9];
    k_pack<<<(64 * 160 + 255) / 256, 256, 0, stream>>>(m1w, w1t[l], 134, 160, 64);
    k_pack<<<(64 * 64 + 255) / 256, 256, 0, stream>>>(m2w, w2t[l], 64, 64, 64);
    k_pack<<<(64 * 128 + 255) / 256, 256, 0, stream>>>(uw1, u1t[l], 128, 128, 64);
    k_pack<<<(64 * 64 + 255) / 256, 256, 0, stream>>>(uw2, u2t[l], 64, 64, 64);
  }

  // ---- input projection ----
  k_input<<<(NN * HH + 255) / 256, 256, 0, stream>>>(x, in_w, in_b, hA);

  // ---- message-passing layers ----
  const int edgeBlocks = (EE / 16 + 7) / 8;   // 62500 tiles / 8 waves
  const int nodeBlocks = (NN + 127) / 128;
  float* hCur = hA;
  float* hNxt = hB;
  for (int l = 0; l < 3; ++l) {
    const int base = 6 + 10 * l;
    const float* ln_b = (const float*)d_in[base + 0];
    const float* ln_g = (const float*)d_in[base + 1];
    const float* m1b  = (const float*)d_in[base + 2];
    const float* m2b  = (const float*)d_in[base + 4];
    const float* u1b  = (const float*)d_in[base + 6];
    const float* u2b  = (const float*)d_in[base + 8];

    hipMemsetAsync(aggr, 0, (size_t)NN * HH * 4, stream);
    k_edge<<<edgeBlocks, 256, 0, stream>>>(hCur, pos, srcIdx, dstIdx,
                                           w1t[l], m1b, w2t[l], m2b, aggr);
    k_update<<<nodeBlocks, 256, 0, stream>>>(hCur, aggr, u1t[l], u1b,
                                             u2t[l], u2b, ln_g, ln_b, hNxt);
    float* tmp = hCur; hCur = hNxt; hNxt = tmp;
  }

  // ---- output projection + segment mean + pool head ----
  hipMemsetAsync(gsum, 0, (size_t)GG * OUTD * 4, stream);
  hipMemsetAsync(gcnt, 0, (size_t)GG * 4, stream);
  k_outproj<<<NN, 128, 0, stream>>>(hCur, batch, out_w, out_b, gsum, gcnt);
  k_pool<<<1, 256, 0, stream>>>(gsum, gcnt, p1w, p1b, p2w, p2b, (float*)d_out);
}
